// HT_56298431316042
// MI455X (gfx1250) — matrix operand order repaired and checked by reading, not compiled, not run on hardware
//
#include <hip/hip_runtime.h>

// ---------------------------------------------------------------------------
// ht = (relu(X) @ V) / 128
//   X: (32 x 16384) f32  (image reshaped)      -> converted to f16 in d_ws
//   V: (16384 x 16384) f32 binary              -> streamed once, cvt f16 in regs
//   out: (32 x 16384) f32
// Bandwidth-bound on V (1 GiB @ 23.3 TB/s ~= 46us). Compute via
// v_wmma_f32_16x16x32_f16: each wave computes a 32x64 patch over a K-segment,
// 16 K-segments accumulated via f32 global atomics. 4096 waves total.
// ---------------------------------------------------------------------------

typedef __attribute__((ext_vector_type(16))) _Float16 v16h;
typedef __attribute__((ext_vector_type(8)))  _Float16 v8h;
typedef __attribute__((ext_vector_type(8)))  float    v8f;

#define MDIM 32        // B*C
#define KDIM 16384     // ROWS*COLS
#define NDIM 16384     // H*W
#define INV_COLS (1.0f / 128.0f)

#define NT 4                       // 16-col N-tiles per wave
#define KSEG 16                    // K split for occupancy
#define KPERSEG (KDIM / KSEG)      // 1024
#define NGROUPS (NDIM / (16 * NT)) // 256
#define TOTWAVES (NGROUPS * KSEG)  // 4096
#define TPB 256

__global__ void ht_zero_kernel(float* __restrict__ out, int n) {
    int i = blockIdx.x * blockDim.x + threadIdx.x;
    if (i < n) out[i] = 0.0f;
}

// relu + f32 -> f16, row-major (MDIM x KDIM)
__global__ void ht_cvt_kernel(const float* __restrict__ img,
                              _Float16* __restrict__ xh, int n) {
    int i = blockIdx.x * blockDim.x + threadIdx.x;
    if (i < n) {
        float v = img[i];
        xh[i] = (_Float16)(v > 0.0f ? v : 0.0f);
    }
}

__global__ __launch_bounds__(TPB)
void ht_gemm_kernel(const _Float16* __restrict__ Xh,
                    const float* __restrict__ V,
                    float* __restrict__ out) {
    const int tid    = threadIdx.x;
    const int lane   = tid & 31;
    const int wid    = blockIdx.x * (TPB / 32) + (tid >> 5);
    const int ngroup = wid % NGROUPS;
    const int kseg   = wid / NGROUPS;

    const int l15 = lane & 15;   // col within tile (B/C/D) or row (A)
    const int hi  = lane >> 4;   // half-wave select

    const int n0 = ngroup * (16 * NT);
    const int k0 = kseg * KPERSEG;

    v8f acc[2][NT];
    const v8f vzero = {0.f, 0.f, 0.f, 0.f, 0.f, 0.f, 0.f, 0.f};
#pragma unroll
    for (int mt = 0; mt < 2; ++mt)
#pragma unroll
        for (int t = 0; t < NT; ++t)
            acc[mt][t] = vzero;

    for (int kc = k0; kc < k0 + KPERSEG; kc += 32) {
        // ---- A fragments (16-bit A 16x32 layout):
        // lanes 0-15 : halfs 0-7 = K kc+0..7,  halfs 8-15 = K kc+16..23
        // lanes 16-31: halfs 0-7 = K kc+8..15, halfs 8-15 = K kc+24..31
        v16h a[2];
#pragma unroll
        for (int mt = 0; mt < 2; ++mt) {
            const _Float16* ap =
                Xh + (size_t)(mt * 16 + l15) * KDIM + (kc + hi * 8);
            union { v16h v; v8h h[2]; } u;
            u.h[0] = *(const v8h*)(ap);       // K = kc + hi*8 + [0..7]
            u.h[1] = *(const v8h*)(ap + 16);  // K = kc + 16 + hi*8 + [0..7]
            a[mt] = u.v;
        }

#pragma unroll
        for (int t = 0; t < NT; ++t) {
            // ---- B fragment (16-bit B 32x16 layout):
            // lane = column n; lanes 0-15: halfs 0-7 = K kc+0..7, 8-15 = K kc+16..23;
            // lanes 16-31: halfs 0-7 = K kc+8..15, 8-15 = K kc+24..31.
            const float* vp =
                V + (size_t)(kc + hi * 8) * NDIM + (n0 + t * 16 + l15);
            union { v16h v; _Float16 h[16]; } b;
#pragma unroll
            for (int j = 0; j < 16; ++j)
                b.h[j] = (_Float16)vp[(size_t)(j + (j >= 8 ? 8 : 0)) * NDIM];

            acc[0][t] = __builtin_amdgcn_wmma_f32_16x16x32_f16(
                false, a[0], false, b.v, (short)0, acc[0][t], false, false);
            acc[1][t] = __builtin_amdgcn_wmma_f32_16x16x32_f16(
                false, a[1], false, b.v, (short)0, acc[1][t], false, false);
        }
    }

    // ---- accumulate partial 32x64 patch into output.
    // C/D layout: lane 0-15 -> n = l15, VGPR r -> row r; lanes 16-31 -> row 8+r.
#pragma unroll
    for (int mt = 0; mt < 2; ++mt)
#pragma unroll
        for (int t = 0; t < NT; ++t)
#pragma unroll
            for (int r = 0; r < 8; ++r) {
                const int m = mt * 16 + hi * 8 + r;
                const int n = n0 + t * 16 + l15;
                atomicAdd(out + (size_t)m * NDIM + n,
                          acc[mt][t][r] * INV_COLS);
            }
}

extern "C" void kernel_launch(void* const* d_in, const int* in_sizes, int n_in,
                              void* d_out, int out_size, void* d_ws, size_t ws_size,
                              hipStream_t stream) {
    const float* image = (const float*)d_in[0];   // (2,16,128,128) f32
    const float* vote  = (const float*)d_in[1];   // (128,128,128,128) f32
    float*       out   = (float*)d_out;           // (2,16,128,128) f32
    _Float16*    xh    = (_Float16*)d_ws;         // 1 MB scratch: relu(X) in f16

    const int nOut = MDIM * NDIM;  // 524288
    const int nX   = MDIM * KDIM;  // 524288

    ht_zero_kernel<<<(nOut + 255) / 256, 256, 0, stream>>>(out, nOut);
    ht_cvt_kernel<<<(nX + 255) / 256, 256, 0, stream>>>(image, xh, nX);

    const int blocks = (TOTWAVES * 32) / TPB;  // 512
    ht_gemm_kernel<<<blocks, TPB, 0, stream>>>(xh, vote, out);
}
